// GraphSAGE_46626164965917
// MI455X (gfx1250) — matrix operand order repaired and checked by reading, not compiled, not run on hardware
//
#include <hip/hip_runtime.h>

#define DF 128  // emb_dim == hidden_dim == 128

typedef __attribute__((ext_vector_type(16))) _Float16 v16h;
typedef __attribute__((ext_vector_type(8)))  float    v8f;

union AFrag { v16h v; _Float16 e[16]; };

__device__ __forceinline__ float wave_sum(float x) {
#pragma unroll
  for (int off = 16; off > 0; off >>= 1) x += __shfl_xor(x, off, 32);
  return x;
}

// ---------------------------------------------------------------------------
// LayerNorm over 128 features. One wave32 per row; lane handles 4 contiguous
// floats. If idx != nullptr, first gathers the embedding row (layer-0 path).
// ---------------------------------------------------------------------------
__global__ void ln_kernel(const int* __restrict__ idx,
                          const float* __restrict__ in,     // [V,128] or [N,128]
                          const float* __restrict__ gamma,
                          const float* __restrict__ beta,
                          float* __restrict__ out, int n) {
  int wid  = (int)(((long long)blockIdx.x * blockDim.x + threadIdx.x) >> 5);
  int lane = threadIdx.x & 31;
  if (wid >= n) return;
  int row = idx ? idx[wid] : wid;
  float4 x = ((const float4*)(in + (size_t)row * DF))[lane];
  float s  = wave_sum(x.x + x.y + x.z + x.w);
  float s2 = wave_sum(x.x * x.x + x.y * x.y + x.z * x.z + x.w * x.w);
  float mean = s * (1.0f / DF);
  float var  = s2 * (1.0f / DF) - mean * mean;
  float inv  = rsqrtf(var + 1e-5f);
  float4 g = ((const float4*)gamma)[lane];
  float4 b = ((const float4*)beta)[lane];
  float4 o;
  o.x = (x.x - mean) * inv * g.x + b.x;
  o.y = (x.y - mean) * inv * g.y + b.y;
  o.z = (x.z - mean) * inv * g.z + b.z;
  o.w = (x.w - mean) * inv * g.w + b.w;
  ((float4*)(out + (size_t)wid * DF))[lane] = o;
}

// ---------------------------------------------------------------------------
// Zero scratch.
// ---------------------------------------------------------------------------
__global__ void zero_kernel(float* __restrict__ p, long long n) {
  long long i = (long long)blockIdx.x * blockDim.x + threadIdx.x;
  if (i < n) p[i] = 0.0f;
}

// ---------------------------------------------------------------------------
// Edge scatter: one wave32 per edge; lane moves 4 floats (coalesced 512B
// gather of x[src], f32 atomic scatter into agg[dst]); lane0 bumps degree.
// ---------------------------------------------------------------------------
__global__ void agg_kernel(const float* __restrict__ x,
                           const int* __restrict__ ei,   // [2,E] row-major
                           float* __restrict__ agg,
                           float* __restrict__ deg, int e) {
  long long gid = (long long)blockIdx.x * blockDim.x + threadIdx.x;
  int edge = (int)(gid >> 5);
  int lane = (int)(gid & 31);
  if (edge >= e) return;
  int s = ei[edge];
  int d = ei[e + edge];
  float4 v = ((const float4*)(x + (size_t)s * DF))[lane];
  float* ap = agg + (size_t)d * DF + lane * 4;
  atomicAdd(ap + 0, v.x);
  atomicAdd(ap + 1, v.y);
  atomicAdd(ap + 2, v.z);
  atomicAdd(ap + 3, v.w);
  if (lane == 0) atomicAdd(deg + d, 1.0f);
}

// ---------------------------------------------------------------------------
// SAGE linear: Y = relu( (agg/max(deg,1)) @ Wl^T + bl + X @ Wr^T )
// Block = 128 threads = 4 waves; each wave owns a 16-row M tile and all 128
// output columns (8x v8f accumulators). K=128 per input, f16 WMMA, f32 acc.
// Weights staged into LDS as f16 once per block.
// ---------------------------------------------------------------------------
__global__ void __launch_bounds__(128)
sage_gemm_kernel(const float* __restrict__ agg,
                 const float* __restrict__ deg,
                 const float* __restrict__ x,
                 const float* __restrict__ wl,
                 const float* __restrict__ bl,
                 const float* __restrict__ wr,
                 float* __restrict__ y, int n) {
  __shared__ __align__(32) _Float16 sW[2][DF * DF];
  int tid = threadIdx.x;
  for (int i = tid; i < DF * DF; i += 128) {
    sW[0][i] = (_Float16)wl[i];
    sW[1][i] = (_Float16)wr[i];
  }
  __syncthreads();

  int wave = tid >> 5;
  int lane = tid & 31;
  int hs   = lane >> 4;      // half-wave select (ISA A/B lane split)
  int lidx = lane & 15;
  int rowBase = blockIdx.x * 64 + wave * 16;
  int m  = rowBase + lidx;
  int mc = m < n ? m : n - 1;          // clamp for safe loads; store guarded

  float invdeg = 1.0f / fmaxf(deg[mc], 1.0f);

  v8f zero = {0.f, 0.f, 0.f, 0.f, 0.f, 0.f, 0.f, 0.f};
  v8f acc[8];
#pragma unroll
  for (int i = 0; i < 8; i++) acc[i] = zero;

#pragma unroll
  for (int ph = 0; ph < 2; ph++) {
    const float* A = ph ? x : agg;
    float scale = ph ? 1.0f : invdeg;
    const _Float16* W = sW[ph];
#pragma unroll
    for (int kt = 0; kt < 4; kt++) {
      // A fragment, contiguous k order (same as B below):
      //   lanes 0-15: e[0..15]=K 0..15
      //   lanes16-31: e[0..15]=K 16..31
      AFrag a;
      const float* ar = A + (size_t)mc * DF + kt * 32;
#pragma unroll
      for (int i = 0; i < 8; i++)
        a.e[i] = (_Float16)(ar[hs * 16 + i] * scale);
#pragma unroll
      for (int i = 0; i < 8; i++)
        a.e[8 + i] = (_Float16)(ar[hs * 16 + 8 + i] * scale);

#pragma unroll
      for (int nt = 0; nt < 8; nt++) {
        // B fragment: col = nt*16+lidx; lanes 0-15 hold K 0..15,
        // lanes 16-31 hold K 16..31 -> 16 contiguous halfs from LDS.
        v16h b = *(const v16h*)(W + (size_t)(nt * 16 + lidx) * DF +
                                kt * 32 + hs * 16);
        acc[nt] = __builtin_amdgcn_wmma_f32_16x16x32_f16(
            false, a.v, false, b, (short)0, acc[nt], false, false);
      }
    }
  }

  // C/D layout: VGPR g -> row = g + 8*hs (within tile), col = lidx.
#pragma unroll
  for (int nt = 0; nt < 8; nt++) {
    int col = nt * 16 + lidx;
    float bias = bl[col];
#pragma unroll
    for (int g2 = 0; g2 < 8; g2++) {
      int row = rowBase + hs * 8 + g2;
      if (row < n) {
        float v = acc[nt][g2] + bias;
        y[(size_t)row * DF + col] = fmaxf(v, 0.0f);
      }
    }
  }
}

// ---------------------------------------------------------------------------
// Global mean pool: one wave32 per node, atomic scatter into per-graph sums.
// ---------------------------------------------------------------------------
__global__ void pool_kernel(const float* __restrict__ x,
                            const int* __restrict__ batch,
                            float* __restrict__ sums,
                            float* __restrict__ cnts, int n) {
  int wid  = (int)(((long long)blockIdx.x * blockDim.x + threadIdx.x) >> 5);
  int lane = threadIdx.x & 31;
  if (wid >= n) return;
  int g = batch[wid];
  float4 v = ((const float4*)(x + (size_t)wid * DF))[lane];
  float* sp = sums + (size_t)g * DF + lane * 4;
  atomicAdd(sp + 0, v.x);
  atomicAdd(sp + 1, v.y);
  atomicAdd(sp + 2, v.z);
  atomicAdd(sp + 3, v.w);
  if (lane == 0) atomicAdd(cnts + g, 1.0f);
}

// ---------------------------------------------------------------------------
// MLP head: one thread per graph (G=256 -> tiny). Weights staged to LDS.
// ---------------------------------------------------------------------------
__global__ void __launch_bounds__(256)
head_kernel(const float* __restrict__ sums, const float* __restrict__ cnts,
            const float* __restrict__ w1, const float* __restrict__ b1,
            const float* __restrict__ w2, const float* __restrict__ b2,
            float* __restrict__ out, int G) {
  __shared__ float sw1[64 * DF];
  __shared__ float sb1[64];
  __shared__ float sw2[2 * 64];
  int tid = threadIdx.x;
  for (int i = tid; i < 64 * DF; i += 256) sw1[i] = w1[i];
  if (tid < 64) sb1[tid] = b1[tid];
  if (tid < 128) sw2[tid] = w2[tid];
  __syncthreads();
  int g = blockIdx.x * blockDim.x + tid;
  if (g >= G) return;
  float inv = 1.0f / fmaxf(cnts[g], 1.0f);
  const float* gs = sums + (size_t)g * DF;
  float o0 = b2[0], o1 = b2[1];
  for (int j = 0; j < 64; j++) {
    float h = sb1[j];
    const float* wj = sw1 + j * DF;
    for (int k = 0; k < DF; k++) h += gs[k] * inv * wj[k];
    h = fmaxf(h, 0.0f);
    o0 += h * sw2[j];
    o1 += h * sw2[64 + j];
  }
  out[g * 2 + 0] = o0;
  out[g * 2 + 1] = o1;
}

// ---------------------------------------------------------------------------
extern "C" void kernel_launch(void* const* d_in, const int* in_sizes, int n_in,
                              void* d_out, int out_size, void* d_ws, size_t ws_size,
                              hipStream_t stream) {
  const int*   x_idx  = (const int*)d_in[0];
  const int*   ei     = (const int*)d_in[1];
  const int*   batch  = (const int*)d_in[2];
  const float* emb    = (const float*)d_in[3];
  const float* ln0_g  = (const float*)d_in[4];
  const float* ln0_b  = (const float*)d_in[5];
  const float* w0_l   = (const float*)d_in[6];
  const float* b0_l   = (const float*)d_in[7];
  const float* w0_r   = (const float*)d_in[8];
  const float* ln1_g  = (const float*)d_in[9];
  const float* ln1_b  = (const float*)d_in[10];
  const float* w1_l   = (const float*)d_in[11];
  const float* b1_l   = (const float*)d_in[12];
  const float* w1_r   = (const float*)d_in[13];
  const float* mlp_w1 = (const float*)d_in[14];
  const float* mlp_b1 = (const float*)d_in[15];
  const float* mlp_w2 = (const float*)d_in[16];
  const float* mlp_b2 = (const float*)d_in[17];
  float* out = (float*)d_out;

  int N = in_sizes[0];
  int E = in_sizes[1] / 2;
  int G = out_size / 2;

  // Workspace layout (agg,deg adjacent and sums,cnts adjacent for one-shot zero)
  char* ws = (char*)d_ws;
  size_t off = 0;
  auto alloc = [&](size_t nf) { float* p = (float*)(ws + off); off += nf * sizeof(float); return p; };
  float* x0   = alloc((size_t)N * DF);   // LN output (both layers)
  float* agg  = alloc((size_t)N * DF);   // edge aggregate
  float* deg  = alloc((size_t)N);        // degrees
  float* y    = alloc((size_t)N * DF);   // conv output (both layers)
  float* sums = alloc((size_t)G * DF);   // pooled sums
  float* cnts = alloc((size_t)G);        // pooled counts
  (void)ws_size; (void)n_in;

  int lnBlk   = (N + 7) / 8;                                   // 8 waves/block
  int aggBlk  = (int)(((long long)E * 32 + 255) / 256);
  int gemmBlk = (N + 63) / 64;                                 // 4 waves x 16 rows
  long long zAgg = (long long)N * DF + N;                      // agg + deg
  long long zPool = (long long)G * DF + G;                     // sums + cnts
  int zAggBlk  = (int)((zAgg + 255) / 256);
  int zPoolBlk = (int)((zPool + 255) / 256);

  // ---- layer 0 ----
  ln_kernel<<<lnBlk, 256, 0, stream>>>(x_idx, emb, ln0_g, ln0_b, x0, N);
  zero_kernel<<<zAggBlk, 256, 0, stream>>>(agg, zAgg);
  agg_kernel<<<aggBlk, 256, 0, stream>>>(x0, ei, agg, deg, E);
  sage_gemm_kernel<<<gemmBlk, 128, 0, stream>>>(agg, deg, x0, w0_l, b0_l, w0_r, y, N);

  // ---- layer 1 ----
  ln_kernel<<<lnBlk, 256, 0, stream>>>(nullptr, y, ln1_g, ln1_b, x0, N);
  zero_kernel<<<zAggBlk, 256, 0, stream>>>(agg, zAgg);
  agg_kernel<<<aggBlk, 256, 0, stream>>>(x0, ei, agg, deg, E);
  sage_gemm_kernel<<<gemmBlk, 128, 0, stream>>>(agg, deg, x0, w1_l, b1_l, w1_r, y, N);

  // ---- pool + head ----
  zero_kernel<<<zPoolBlk, 256, 0, stream>>>(sums, zPool);
  pool_kernel<<<lnBlk, 256, 0, stream>>>(y, batch, sums, cnts, N);
  head_kernel<<<(G + 255) / 256, 256, 0, stream>>>(sums, cnts, mlp_w1, mlp_b1,
                                                   mlp_w2, mlp_b2, out, G);
}
